// Net_14405320311622
// MI455X (gfx1250) — compile-verified, matrix operand order checked by reading, not run on hardware
//
#include <hip/hip_runtime.h>

// ---------------------------------------------------------------------------
// MI455X / gfx1250 PointConv GNN.  All heavy matmuls via v_wmma_f32_16x16x32_f16.
// ---------------------------------------------------------------------------

typedef _Float16 half16 __attribute__((ext_vector_type(16)));
typedef _Float16 half8  __attribute__((ext_vector_type(8)));
typedef float    v8f    __attribute__((ext_vector_type(8)));

#define NNODES 50000
#define NEDGES 250000
#define NBATCH 8
#define HEAD_IN 896

__device__ __forceinline__ half16 mk16(half8 lo, half8 hi) {
  half16 r;
#pragma unroll
  for (int i = 0; i < 8; ++i) { r[i] = lo[i]; r[i + 8] = hi[i]; }
  return r;
}

// K index held by (lane, slot) in a 16-bit 16x32 A / 32x16 B WMMA fragment.
// slot = r*2+p  (VGPR r, packed half p).  Matches CDNA5 ISA 7.12.2 layout.
__device__ __host__ __forceinline__ int frag_k(int lane, int slot) {
  int base0 = (lane < 16) ? 0 : 8;
  int r = slot >> 1, p = slot & 1;
  return (r < 4) ? (base0 + 2 * r + p) : (16 + base0 + 2 * (r - 4) + p);
}

// ---------------------------------------------------------------------------
// Convert a row-major f32 weight matrix W[Kreal x Ncols] into f16 WMMA B-matrix
// fragments.  Fragment (kb, nb) occupies 512 contiguous halves; lane-local data
// is 16 contiguous halves so a wave loads its B operand with 2x b128.
// ---------------------------------------------------------------------------
__global__ void build_frags(const float* __restrict__ W, _Float16* __restrict__ out,
                            int Kreal, int Kpad, int Ncols) {
  int kblocks = Kpad >> 5;
  int total = kblocks * (Ncols >> 4) * 512;
  for (int i = blockIdx.x * blockDim.x + threadIdx.x; i < total;
       i += gridDim.x * blockDim.x) {
    int frag = i >> 9;
    int rem  = i & 511;
    int lane = rem >> 4;
    int slot = rem & 15;
    int nb = frag / kblocks;
    int kb = frag - nb * kblocks;
    int n = nb * 16 + (lane & 15);
    int k = kb * 32 + frag_k(lane, slot);
    out[i] = (_Float16)((k < Kreal) ? W[k * Ncols + n] : 0.0f);
  }
}

// ---------------------------------------------------------------------------
// One PointConv layer:
//   per edge e: h = relu([x[src]||pos[src]-pos[dst]] @ W1 + b1) @ W2 + b2
//   out[dst]   = max(0, max_e h)   (bit-pattern int atomicMax, out pre-zeroed)
// Block = 256 threads = 8 waves; each wave owns 16 edges (one WMMA M-tile).
// ---------------------------------------------------------------------------
template <int DX, int KPAD, int DH, int DOUT>
__global__ __launch_bounds__(256)
void point_conv_kernel(const float* __restrict__ x,     // [N, DX] (null if DX==0)
                       const float* __restrict__ pos,   // [N, 3]
                       const int* __restrict__ eidx,    // [2, E] flat
                       const _Float16* __restrict__ w1f,
                       const _Float16* __restrict__ w2f,
                       const float* __restrict__ b1,
                       const float* __restrict__ b2,
                       float* __restrict__ out,         // [N, DOUT] pre-zeroed
                       int E) {
  extern __shared__ char smem[];
  _Float16* Ah = (_Float16*)smem;                                  // [128][KPAD]
  _Float16* Hh = (_Float16*)(smem + 128 * KPAD * 2);               // [128][DH]
  int* sSrc = (int*)(smem + 128 * KPAD * 2 + 128 * DH * 2);        // [128]
  int* sDst = sSrc + 128;

  const int tid = threadIdx.x;
  const int ebase = blockIdx.x * 128;

  // ---- phase 0: edge indices ------------------------------------------------
  if (tid < 128) {
    int e = ebase + tid;
    int ec = (e < E) ? e : (E - 1);
    sSrc[tid] = eidx[ec];
    sDst[tid] = eidx[E + ec];
  }
  __syncthreads();

  // ---- phase 1: build f16 edge-feature tile (A matrix) ----------------------
  const int TOT = 128 * KPAD;   // multiple of 256 (KPAD even)
  for (int idx = tid; idx < TOT; idx += 256) {
    int e = idx / KPAD;
    int f = idx - e * KPAD;
    int s = sSrc[e];
    float v;
    if constexpr (DX > 0) {
      if (f < DX)            v = x[s * DX + f];
      else if (f < DX + 3)   v = pos[s * 3 + (f - DX)] - pos[sDst[e] * 3 + (f - DX)];
      else                   v = 0.0f;
    } else {
      if (f < 3)             v = pos[s * 3 + f] - pos[sDst[e] * 3 + f];
      else                   v = 0.0f;
    }
    Ah[idx] = (_Float16)v;
  }
  __syncthreads();

  const int wave  = tid >> 5;
  const int lane  = tid & 31;
  const int col   = lane & 15;                 // N-column / M-row selector
  const int base0 = (lane < 16) ? 0 : 8;
  const int rbase = (lane < 16) ? 0 : 8;       // C/D row base for this lane

  // ---- phase 2: GEMM1  h = relu(A @ W1 + b1) --------------------------------
  constexpr int KB1 = KPAD / 32;
  constexpr int NB1 = DH / 16;
  const _Float16* Arow = Ah + (wave * 16 + col) * KPAD;

  for (int nb = 0; nb < NB1; ++nb) {
    float bias = b1[nb * 16 + col];
    v8f acc;
#pragma unroll
    for (int i = 0; i < 8; ++i) acc[i] = bias;
    const _Float16* wf = w1f + (size_t)nb * KB1 * 512;
#pragma unroll
    for (int kb = 0; kb < KB1; ++kb) {
      half8 alo = *(const half8*)(Arow + kb * 32 + base0);
      half8 ahi = *(const half8*)(Arow + kb * 32 + 16 + base0);
      half16 a = mk16(alo, ahi);
      half16 b = *(const half16*)(wf + kb * 512 + lane * 16);
      acc = __builtin_amdgcn_wmma_f32_16x16x32_f16(false, a, false, b,
                                                   (short)0, acc, false, false);
    }
    // relu + stage H tile (f16) for GEMM2
#pragma unroll
    for (int r = 0; r < 8; ++r) {
      float v = acc[r] > 0.0f ? acc[r] : 0.0f;
      Hh[(wave * 16 + rbase + r) * DH + nb * 16 + col] = (_Float16)v;
    }
  }
  __syncthreads();

  // ---- phase 3: GEMM2  o = H @ W2 + b2, scatter atomic max ------------------
  constexpr int KB2 = DH / 32;
  constexpr int NB2 = DOUT / 16;
  const _Float16* Hrow = Hh + (wave * 16 + col) * DH;

  for (int nb = 0; nb < NB2; ++nb) {
    float bias = b2[nb * 16 + col];
    v8f acc;
#pragma unroll
    for (int i = 0; i < 8; ++i) acc[i] = bias;
    const _Float16* wf = w2f + (size_t)nb * KB2 * 512;
#pragma unroll
    for (int kb = 0; kb < KB2; ++kb) {
      half8 alo = *(const half8*)(Hrow + kb * 32 + base0);
      half8 ahi = *(const half8*)(Hrow + kb * 32 + 16 + base0);
      half16 a = mk16(alo, ahi);
      half16 b = *(const half16*)(wf + kb * 512 + lane * 16);
      acc = __builtin_amdgcn_wmma_f32_16x16x32_f16(false, a, false, b,
                                                   (short)0, acc, false, false);
    }
    int n = nb * 16 + col;
#pragma unroll
    for (int r = 0; r < 8; ++r) {
      int eloc = wave * 16 + rbase + r;
      if (ebase + eloc < E) {
        int node = sDst[eloc];
        // relu(max(...)) via signed-int max on float bits against 0-initialized out.
        atomicMax((int*)(out + (size_t)node * DOUT + n), __float_as_int(acc[r]));
      }
    }
  }
}

// ---------------------------------------------------------------------------
// Global max pool over sorted batch ids into g[8][896] (bit-pattern int max).
// ---------------------------------------------------------------------------
__global__ void pool_kernel(const float* __restrict__ x0, const float* __restrict__ x1,
                            const float* __restrict__ x2, const float* __restrict__ x3,
                            const float* __restrict__ x4, const float* __restrict__ x5,
                            const int* __restrict__ batch, int* __restrict__ g, int N) {
  int f = blockIdx.x * blockDim.x + threadIdx.x;
  if (f >= HEAD_IN) return;
  const float* src; int dim, off;
  if      (f < 64)  { src = x0; dim = 64;  off = f;       }
  else if (f < 128) { src = x1; dim = 64;  off = f - 64;  }
  else if (f < 256) { src = x2; dim = 128; off = f - 128; }
  else if (f < 384) { src = x3; dim = 128; off = f - 256; }
  else if (f < 640) { src = x4; dim = 256; off = f - 384; }
  else              { src = x5; dim = 256; off = f - 640; }
  int n0 = blockIdx.y * 256;
  int n1 = n0 + 256; if (n1 > N) n1 = N;
  if (n0 >= N) return;
  int curB = batch[n0];
  float curMax = 0.0f;  // features are >= 0
  for (int n = n0; n < n1; ++n) {
    int b = batch[n];
    if (b != curB) {
      atomicMax(&g[curB * HEAD_IN + f], __float_as_int(curMax));
      curB = b; curMax = 0.0f;
    }
    float v = src[(size_t)n * dim + off];
    curMax = fmaxf(curMax, v);
  }
  atomicMax(&g[curB * HEAD_IN + f], __float_as_int(curMax));
}

// ---------------------------------------------------------------------------
// Head MLP on 8 rows: 896 -> 128 -> 64 -> 1.  Single block, trivial FLOPs.
// ---------------------------------------------------------------------------
__global__ void head_kernel(const float* __restrict__ g,
                            const float* __restrict__ W1, const float* __restrict__ B1,
                            const float* __restrict__ W2, const float* __restrict__ B2,
                            const float* __restrict__ W3, const float* __restrict__ B3,
                            float* __restrict__ out) {
  __shared__ float t1[NBATCH * 128];
  __shared__ float t2[NBATCH * 64];
  int tid = threadIdx.x;  // 128 threads
  if (tid < 128) {
    for (int r = 0; r < NBATCH; ++r) {
      float acc = B1[tid];
      for (int k = 0; k < HEAD_IN; ++k) acc += g[r * HEAD_IN + k] * W1[k * 128 + tid];
      t1[r * 128 + tid] = fmaxf(acc, 0.0f);
    }
  }
  __syncthreads();
  if (tid < 64) {
    for (int r = 0; r < NBATCH; ++r) {
      float acc = B2[tid];
      for (int k = 0; k < 128; ++k) acc += t1[r * 128 + k] * W2[k * 64 + tid];
      t2[r * 64 + tid] = fmaxf(acc, 0.0f);
    }
  }
  __syncthreads();
  if (tid < NBATCH) {
    float acc = B3[0];
    for (int k = 0; k < 64; ++k) acc += t2[tid * 64 + k] * W3[k];
    out[tid] = acc;
  }
}

// ---------------------------------------------------------------------------
// Launch
// ---------------------------------------------------------------------------
extern "C" void kernel_launch(void* const* d_in, const int* in_sizes, int n_in,
                              void* d_out, int out_size, void* d_ws, size_t ws_size,
                              hipStream_t stream) {
  const float* pos  = (const float*)d_in[0];
  // batch / edge indices: harness delivers integers as int32
  const int* batch  = (const int*)d_in[1];
  const int* eit    = (const int*)d_in[2];
  const int* eiz    = (const int*)d_in[3];
  // conv params: layer l -> W1, b1, W2, b2 at 4 + 4l
  const float* W1p[6], *b1p[6], *W2p[6], *b2p[6];
  for (int l = 0; l < 6; ++l) {
    W1p[l] = (const float*)d_in[4 + 4 * l + 0];
    b1p[l] = (const float*)d_in[4 + 4 * l + 1];
    W2p[l] = (const float*)d_in[4 + 4 * l + 2];
    b2p[l] = (const float*)d_in[4 + 4 * l + 3];
  }
  const float* lin1W = (const float*)d_in[28]; const float* lin1b = (const float*)d_in[29];
  const float* lin2W = (const float*)d_in[30]; const float* lin2b = (const float*)d_in[31];
  const float* lin3W = (const float*)d_in[32]; const float* lin3b = (const float*)d_in[33];

  // layer geometry
  const int DXs[6]   = {0, 64, 64, 128, 128, 256};
  const int KPADs[6] = {32, 96, 96, 160, 160, 288};
  const int KREs[6]  = {3, 67, 67, 131, 131, 259};
  const int DHs[6]   = {64, 64, 128, 128, 256, 256};
  const int DOs[6]   = {64, 64, 128, 128, 256, 256};

  // workspace layout
  char* ws = (char*)d_ws;
  size_t off = 0;
  float* xb[6];
  for (int l = 0; l < 6; ++l) {
    xb[l] = (float*)(ws + off);
    off += (size_t)NNODES * DOs[l] * sizeof(float);
  }
  int* g = (int*)(ws + off);
  off += (size_t)NBATCH * HEAD_IN * sizeof(float);
  size_t zero_bytes = off;                      // everything above needs zero init
  off = (off + 63) & ~(size_t)63;
  _Float16* w1f[6]; _Float16* w2f[6];
  for (int l = 0; l < 6; ++l) {
    w1f[l] = (_Float16*)(ws + off); off += (size_t)KPADs[l] * DHs[l] * 2; off = (off + 63) & ~(size_t)63;
    w2f[l] = (_Float16*)(ws + off); off += (size_t)DHs[l] * DOs[l] * 2;   off = (off + 63) & ~(size_t)63;
  }

  // 1) zero node-feature buffers + pooled g (int-max-on-float-bits needs 0 init)
  hipMemsetAsync(d_ws, 0, zero_bytes, stream);

  // 2) weight fragment conversion (cheap; deterministic each call)
  for (int l = 0; l < 6; ++l) {
    int t1 = KPADs[l] * DHs[l];
    build_frags<<<(t1 + 255) / 256, 256, 0, stream>>>(W1p[l], w1f[l], KREs[l], KPADs[l], DHs[l]);
    int t2 = DHs[l] * DOs[l];
    build_frags<<<(t2 + 255) / 256, 256, 0, stream>>>(W2p[l], w2f[l], DHs[l], DHs[l], DOs[l]);
  }

  // 3) six PointConv layers (alternating edge sets)
  const int E = NEDGES;
  const int grid = (E + 127) / 128;
  auto lds = [](int KPAD, int DH) { return (size_t)(128 * KPAD + 128 * DH) * 2 + 1024; };

  point_conv_kernel<0, 32, 64, 64><<<grid, 256, lds(32, 64), stream>>>(
      nullptr, pos, eit, w1f[0], w2f[0], b1p[0], b2p[0], xb[0], E);
  point_conv_kernel<64, 96, 64, 64><<<grid, 256, lds(96, 64), stream>>>(
      xb[0], pos, eiz, w1f[1], w2f[1], b1p[1], b2p[1], xb[1], E);
  point_conv_kernel<64, 96, 128, 128><<<grid, 256, lds(96, 128), stream>>>(
      xb[1], pos, eit, w1f[2], w2f[2], b1p[2], b2p[2], xb[2], E);
  point_conv_kernel<128, 160, 128, 128><<<grid, 256, lds(160, 128), stream>>>(
      xb[2], pos, eiz, w1f[3], w2f[3], b1p[3], b2p[3], xb[3], E);
  point_conv_kernel<128, 160, 256, 256><<<grid, 256, lds(160, 256), stream>>>(
      xb[3], pos, eit, w1f[4], w2f[4], b1p[4], b2p[4], xb[4], E);
  point_conv_kernel<256, 288, 256, 256><<<grid, 256, lds(288, 256), stream>>>(
      xb[4], pos, eiz, w1f[5], w2f[5], b1p[5], b2p[5], xb[5], E);

  // 4) global max pool (sorted batch -> register-accumulated, few atomics)
  dim3 pgrid((HEAD_IN + 127) / 128, (NNODES + 255) / 256);
  pool_kernel<<<pgrid, 128, 0, stream>>>(xb[0], xb[1], xb[2], xb[3], xb[4], xb[5],
                                         batch, g, NNODES);

  // 5) head MLP -> d_out[8]
  head_kernel<<<1, 128, 0, stream>>>((const float*)g, lin1W, lin1b, lin2W, lin2b,
                                     lin3W, lin3b, (float*)d_out);
}
